// DTLN_P2_stateful_36807869727279
// MI455X (gfx1250) — matrix operand order repaired and checked by reading, not compiled, not run on hardware
//
#include <hip/hip_runtime.h>
#include <hip/hip_bf16.h>

// Problem constants
#define B_  32
#define F_  512
#define T_  2000
#define E_  256
#define H_  128
#define G4_ 512   // 4*H

typedef __attribute__((ext_vector_type(16))) __bf16 bf16x16;
typedef __attribute__((ext_vector_type(8)))  float  f32x8;

__device__ __forceinline__ f32x8 wmma_bf16(bf16x16 a, bf16x16 b, f32x8 c) {
    // D = A(16x32) * B(32x16) + C(16x16), f32 accumulate
    return __builtin_amdgcn_wmma_f32_16x16x32_bf16(false, a, false, b,
                                                   (short)0, c, false, false);
}

// A-matrix (16x32, 16-bit) lane element -> K index (contiguous order k = 16*g + i, same order as the B operand)
__device__ __forceinline__ int a_kof(int i, int g) { return i + g * 16; }

// Inverse map: (b,u) -> linear index in fragment-ordered h staging buffer
__device__ __forceinline__ int hstage_idx(int b, int u) {
    int mt = b >> 4, mm = b & 15;
    int ks = u >> 5, kk = u & 31;
    int g = kk >> 4;
    int i = kk & 15;
    return ((mt * 4 + ks) * 32 + g * 16 + mm) * 16 + i;
}

__device__ __forceinline__ float sigmf(float x) { return 1.f / (1.f + __expf(-x)); }

// ---------------------------------------------------------------------------
// Kernel 1: enc = einsum('ef,bft->bte') + LayerNorm.  16 waves / block, each
// wave owns one 16-wide e-tile; block owns a 16-row t-tile of one batch b.
// ---------------------------------------------------------------------------
__global__ void k_enc_ln(const float* __restrict__ y1, const float* __restrict__ enc_w,
                         const float* __restrict__ gamma, const float* __restrict__ beta,
                         float* __restrict__ enc, __bf16* __restrict__ normed)
{
    __shared__ float tile[16][E_];
    __shared__ float s_mean[16], s_rstd[16];
    int tid = threadIdx.x, w = tid >> 5, lane = tid & 31;
    int lm = lane & 15, lg = lane >> 4;
    int blk = blockIdx.x;
    int b = blk / (T_ / 16), tt = blk % (T_ / 16), t0 = tt * 16;
    const float* ybase = y1 + (size_t)b * F_ * T_ + t0;     // + k*T_ + m
    int e0 = w * 16;
    f32x8 acc = {};
    for (int ks = 0; ks < F_ / 32; ++ks) {
        int k0 = ks * 32;
        bf16x16 a, bm;
#pragma unroll
        for (int i = 0; i < 16; ++i)
            a[i] = (__bf16)ybase[(size_t)(k0 + a_kof(i, lg)) * T_ + lm];
        const float* wr = enc_w + (size_t)(e0 + lm) * F_ + k0 + lg * 16;
#pragma unroll
        for (int i = 0; i < 16; ++i) bm[i] = (__bf16)wr[i];
        acc = wmma_bf16(a, bm, acc);
    }
#pragma unroll
    for (int r = 0; r < 8; ++r) tile[r + lg * 8][e0 + lm] = acc[r];
    __syncthreads();
    {   // wave w reduces row w over E
        int row = w;
        float s = 0.f, sq = 0.f;
        for (int e = lane; e < E_; e += 32) { float v = tile[row][e]; s += v; sq += v * v; }
#pragma unroll
        for (int off = 16; off; off >>= 1) {
            s  += __shfl_xor(s,  off, 32);
            sq += __shfl_xor(sq, off, 32);
        }
        if (lane == 0) {
            float mean = s * (1.f / E_);
            float var  = sq * (1.f / E_) - mean * mean;
            s_mean[row] = mean;
            s_rstd[row] = rsqrtf(var + 1e-7f);
        }
    }
    __syncthreads();
    size_t rowbase = (size_t)b * T_ + t0;
    for (int j = 0; j < 8; ++j) {
        int idx = j * 512 + tid; int mm = idx >> 8, e = idx & 255;
        float v = tile[mm][e];
        size_t o = (rowbase + mm) * E_ + e;
        enc[o] = v;
        normed[o] = (__bf16)((v - s_mean[mm]) * s_rstd[mm] * gamma[e] + beta[e]);
    }
}

// ---------------------------------------------------------------------------
// Kernel 2: G1 = normed @ Wih1^T + (bih1+bhh1)   (non-recurrent LSTM1 input)
// ---------------------------------------------------------------------------
__global__ void k_g1(const __bf16* __restrict__ normed, const float* __restrict__ Wih1,
                     const float* __restrict__ bih1, const float* __restrict__ bhh1,
                     float* __restrict__ G1)
{
    int tid = threadIdx.x, w = tid >> 5, lane = tid & 31;
    int lm = lane & 15, lg = lane >> 4;
    size_t row0 = (size_t)blockIdx.x * 16;
    int nt = blockIdx.y * 4 + w, n0 = nt * 16;
    f32x8 acc = {};
    for (int ks = 0; ks < E_ / 32; ++ks) {
        int k0 = ks * 32;
        bf16x16 a, bm;
        const __bf16* ar = normed + (row0 + lm) * E_ + k0;
#pragma unroll
        for (int i = 0; i < 16; ++i) a[i] = ar[a_kof(i, lg)];
        const float* wr = Wih1 + (size_t)(n0 + lm) * E_ + k0 + lg * 16;
#pragma unroll
        for (int i = 0; i < 16; ++i) bm[i] = (__bf16)wr[i];
        acc = wmma_bf16(a, bm, acc);
    }
    int n = n0 + lm;
    float bsum = bih1[n] + bhh1[n];
#pragma unroll
    for (int r = 0; r < 8; ++r)
        G1[(row0 + r + lg * 8) * G4_ + n] = acc[r] + bsum;
}

// ---------------------------------------------------------------------------
// Kernel 3: pre-pack Wih2 into WMMA B-fragment order (bf16) for L2 streaming
// ---------------------------------------------------------------------------
__global__ void k_pack_wih2(const float* __restrict__ Wih2, __bf16* __restrict__ out)
{
    const int tot = G4_ * H_;
    for (int d = blockIdx.x * blockDim.x + threadIdx.x; d < tot; d += gridDim.x * blockDim.x) {
        int i = d & 15, L = (d >> 4) & 31, frag = d >> 9;
        int ks = frag & 3, nt = frag >> 2;
        int n = nt * 16 + (L & 15), g = L >> 4;
        int k = ks * 32 + i + g * 16;
        out[d] = (__bf16)Wih2[n * H_ + k];
    }
}

// ---------------------------------------------------------------------------
// Kernel 4: persistent 2-layer LSTM scan.  One workgroup (512 thr = 16 waves).
// LDS (dynamic, 311296 B < 320 KB/WGP): Whh1,Whh2 bf16 fragments (2x128 KB),
// 32 KB z buffer, 2x8 KB h staging.  Cell state lives in registers.
// ---------------------------------------------------------------------------
__global__ void k_lstm(const float* __restrict__ G1, const __bf16* __restrict__ wih2f,
                       const float* __restrict__ Whh1, const float* __restrict__ Whh2,
                       const float* __restrict__ bih2, const float* __restrict__ bhh2,
                       const float* __restrict__ in_state, __bf16* __restrict__ ys2,
                       float* __restrict__ out_state)
{
    extern __shared__ char smem[];
    __bf16* whh1 = (__bf16*)(smem);
    __bf16* whh2 = (__bf16*)(smem + 131072);
    __bf16* zbuf = (__bf16*)(smem + 262144);
    __bf16* h1s  = (__bf16*)(smem + 294912);
    __bf16* h2s  = (__bf16*)(smem + 303104);

    int tid = threadIdx.x, w = tid >> 5, lane = tid & 31;
    int lm = lane & 15, lg = lane >> 4;

    // pack Whh1/Whh2 -> LDS fragment order (once)
    for (int d = tid; d < G4_ * H_; d += 512) {
        int i = d & 15, L = (d >> 4) & 31, frag = d >> 9;
        int ks = frag & 3, nt = frag >> 2;
        int n = nt * 16 + (L & 15), g = L >> 4;
        int k = ks * 32 + i + g * 16;
        whh1[d] = (__bf16)Whh1[n * H_ + k];
        whh2[d] = (__bf16)Whh2[n * H_ + k];
    }

    // per-thread state: 8 (b,u) pairs each
    float c1r[8], c2r[8], b2i[8], b2f[8], b2g[8], b2o[8];
    int pb[8], pu[8];
#pragma unroll
    for (int j = 0; j < 8; ++j) {
        int p = j * 512 + tid; int b = p >> 7, u = p & 127;
        pb[j] = b; pu[j] = u;
        float h1 = in_state[((size_t)(b) * H_ + u) * 2 + 0];
        c1r[j]   = in_state[((size_t)(b) * H_ + u) * 2 + 1];
        float h2 = in_state[((size_t)(B_ + b) * H_ + u) * 2 + 0];
        c2r[j]   = in_state[((size_t)(B_ + b) * H_ + u) * 2 + 1];
        int hi = hstage_idx(b, u);
        h1s[hi] = (__bf16)h1;
        h2s[hi] = (__bf16)h2;
        b2i[j] = bih2[u]       + bhh2[u];
        b2f[j] = bih2[128 + u] + bhh2[128 + u];
        b2g[j] = bih2[256 + u] + bhh2[256 + u];
        b2o[j] = bih2[384 + u] + bhh2[384 + u];
    }
    __syncthreads();

    for (int t = 0; t < T_; ++t) {
        // ---- layer 1: z1 = h1 @ Whh1^T + G1[:,t,:]
#pragma unroll
        for (int q = 0; q < 4; ++q) {
            int tau = q * 16 + w; int mt = tau >> 5, nt = tau & 31;
            f32x8 acc = {};
#pragma unroll
            for (int ks = 0; ks < 4; ++ks) {
                bf16x16 a  = *(const bf16x16*)&h1s [((mt * 4 + ks) * 32 + lane) * 16];
                bf16x16 bm = *(const bf16x16*)&whh1[((nt * 4 + ks) * 32 + lane) * 16];
                acc = wmma_bf16(a, bm, acc);
            }
            int n = nt * 16 + lm;
            if (t + 1 < T_)  // warm L2/L0 for next step's G1 row
                __builtin_prefetch(&G1[((size_t)(mt * 16 + lg * 8) * T_ + t + 1) * G4_ + n], 0, 1);
#pragma unroll
            for (int r = 0; r < 8; ++r) {
                int b = mt * 16 + r + lg * 8;
                float z = acc[r] + G1[((size_t)b * T_ + t) * G4_ + n];
                zbuf[b * G4_ + n] = (__bf16)z;
            }
        }
        __syncthreads();
        // ---- layer 1 gates
#pragma unroll
        for (int j = 0; j < 8; ++j) {
            int b = pb[j], u = pu[j];
            const __bf16* zr = zbuf + b * G4_;
            float zi = (float)zr[u],        zf = (float)zr[128 + u];
            float zg = (float)zr[256 + u],  zo = (float)zr[384 + u];
            float c = sigmf(zf) * c1r[j] + sigmf(zi) * tanhf(zg);
            c1r[j] = c;
            h1s[hstage_idx(b, u)] = (__bf16)(sigmf(zo) * tanhf(c));
        }
        __syncthreads();
        // ---- layer 2: z2 = h1_t @ Wih2^T (L2 stream) + h2 @ Whh2^T (LDS) + b2
#pragma unroll
        for (int q = 0; q < 4; ++q) {
            int tau = q * 16 + w; int mt = tau >> 5, nt = tau & 31;
            f32x8 acc = {};
#pragma unroll
            for (int ks = 0; ks < 4; ++ks) {
                bf16x16 a  = *(const bf16x16*)&h1s  [((mt * 4 + ks) * 32 + lane) * 16];
                bf16x16 bm = *(const bf16x16*)&wih2f[((nt * 4 + ks) * 32 + lane) * 16];
                acc = wmma_bf16(a, bm, acc);
            }
#pragma unroll
            for (int ks = 0; ks < 4; ++ks) {
                bf16x16 a  = *(const bf16x16*)&h2s [((mt * 4 + ks) * 32 + lane) * 16];
                bf16x16 bm = *(const bf16x16*)&whh2[((nt * 4 + ks) * 32 + lane) * 16];
                acc = wmma_bf16(a, bm, acc);
            }
            int n = nt * 16 + lm;
#pragma unroll
            for (int r = 0; r < 8; ++r) {
                int b = mt * 16 + r + lg * 8;
                zbuf[b * G4_ + n] = (__bf16)acc[r];
            }
        }
        __syncthreads();
        // ---- layer 2 gates + ys2 emit
#pragma unroll
        for (int j = 0; j < 8; ++j) {
            int b = pb[j], u = pu[j];
            const __bf16* zr = zbuf + b * G4_;
            float zi = (float)zr[u]       + b2i[j], zf = (float)zr[128 + u] + b2f[j];
            float zg = (float)zr[256 + u] + b2g[j], zo = (float)zr[384 + u] + b2o[j];
            float c = sigmf(zf) * c2r[j] + sigmf(zi) * tanhf(zg);
            c2r[j] = c;
            float h = sigmf(zo) * tanhf(c);
            h2s[hstage_idx(b, u)] = (__bf16)h;
            ys2[((size_t)b * T_ + t) * H_ + u] = (__bf16)h;
        }
        __syncthreads();
    }
    // final states: (2,B,H,2) = [layer][b][u][h|c]
#pragma unroll
    for (int j = 0; j < 8; ++j) {
        int b = pb[j], u = pu[j]; int hi = hstage_idx(b, u);
        out_state[((size_t)(b) * H_ + u) * 2 + 0]      = (float)h1s[hi];
        out_state[((size_t)(b) * H_ + u) * 2 + 1]      = c1r[j];
        out_state[((size_t)(B_ + b) * H_ + u) * 2 + 0] = (float)h2s[hi];
        out_state[((size_t)(B_ + b) * H_ + u) * 2 + 1] = c2r[j];
    }
}

// ---------------------------------------------------------------------------
// Kernel 5: mask = sigmoid(ys2 @ Wd^T + bd); est = mask * enc   (bf16 out)
// ---------------------------------------------------------------------------
__global__ void k_mask(const __bf16* __restrict__ ys2, const float* __restrict__ Wd,
                       const float* __restrict__ bd, const float* __restrict__ enc,
                       __bf16* __restrict__ est)
{
    int tid = threadIdx.x, w = tid >> 5, lane = tid & 31;
    int lm = lane & 15, lg = lane >> 4;
    size_t row0 = (size_t)blockIdx.x * 16;
    int nt = blockIdx.y * 4 + w, n0 = nt * 16;
    f32x8 acc = {};
    for (int ks = 0; ks < H_ / 32; ++ks) {
        int k0 = ks * 32;
        bf16x16 a, bm;
        const __bf16* ar = ys2 + (row0 + lm) * H_ + k0;
#pragma unroll
        for (int i = 0; i < 16; ++i) a[i] = ar[a_kof(i, lg)];
        const float* wr = Wd + (size_t)(n0 + lm) * H_ + k0 + lg * 16;
#pragma unroll
        for (int i = 0; i < 16; ++i) bm[i] = (__bf16)wr[i];
        acc = wmma_bf16(a, bm, acc);
    }
    int e = n0 + lm;
    float bde = bd[e];
#pragma unroll
    for (int r = 0; r < 8; ++r) {
        size_t row = row0 + r + lg * 8;
        float mask = sigmf(acc[r] + bde);
        est[row * E_ + e] = (__bf16)(mask * enc[row * E_ + e]);
    }
}

// ---------------------------------------------------------------------------
// Kernel 6: decoded = einsum('fe,bte->bft', dec_w, est)
// ---------------------------------------------------------------------------
__global__ void k_dec(const float* __restrict__ dec_w, const __bf16* __restrict__ est,
                      float* __restrict__ outp)
{
    int tid = threadIdx.x, w = tid >> 5, lane = tid & 31;
    int lm = lane & 15, lg = lane >> 4;
    int blk = blockIdx.x;
    int b = blk / (T_ / 16), tt = blk % (T_ / 16), t0 = tt * 16;
    int ft = blockIdx.y * 4 + w, f0 = ft * 16;
    f32x8 acc = {};
    const __bf16* brow = est + ((size_t)b * T_ + t0 + lm) * E_;   // n = lm (t)
    for (int ks = 0; ks < E_ / 32; ++ks) {
        int k0 = ks * 32;
        bf16x16 a;
        const float* ar = dec_w + (size_t)(f0 + lm) * E_ + k0;
#pragma unroll
        for (int i = 0; i < 16; ++i) a[i] = (__bf16)ar[a_kof(i, lg)];
        bf16x16 bm = *(const bf16x16*)&brow[k0 + lg * 16];
        acc = wmma_bf16(a, bm, acc);
    }
    int t = t0 + lm;
#pragma unroll
    for (int r = 0; r < 8; ++r) {
        int f = f0 + r + lg * 8;
        outp[(size_t)b * F_ * T_ + (size_t)f * T_ + t] = acc[r];
    }
}

// ---------------------------------------------------------------------------
extern "C" void kernel_launch(void* const* d_in, const int* in_sizes, int n_in,
                              void* d_out, int out_size, void* d_ws, size_t ws_size,
                              hipStream_t stream) {
    const float* y1     = (const float*)d_in[0];
    const float* inst2  = (const float*)d_in[1];
    const float* enc_w  = (const float*)d_in[2];
    const float* gamma  = (const float*)d_in[3];
    const float* beta   = (const float*)d_in[4];
    const float* Wih1   = (const float*)d_in[5];
    const float* Whh1   = (const float*)d_in[6];
    const float* bih1   = (const float*)d_in[7];
    const float* bhh1   = (const float*)d_in[8];
    const float* Wih2   = (const float*)d_in[9];
    const float* Whh2   = (const float*)d_in[10];
    const float* bih2   = (const float*)d_in[11];
    const float* bhh2   = (const float*)d_in[12];
    const float* Wd     = (const float*)d_in[13];
    const float* bd     = (const float*)d_in[14];
    const float* dec_w  = (const float*)d_in[15];

    float*  decoded   = (float*)d_out;
    float*  out_state = decoded + (size_t)B_ * F_ * T_;

    char* ws = (char*)d_ws;
    float*  enc    = (float*)(ws);                                       // 65,536,000 B
    __bf16* normed = (__bf16*)(ws + 65536000ULL);                        // 32,768,000 B
    float*  G1     = (float*)(ws + 98304000ULL);                         // 131,072,000 B
    __bf16* ys2    = (__bf16*)(ws + 229376000ULL);                       // 16,384,000 B
    __bf16* est    = (__bf16*)(ws + 245760000ULL);                       // 32,768,000 B
    __bf16* wih2f  = (__bf16*)(ws + 278528000ULL);                       // 131,072 B

    const int rowBlocks = B_ * T_ / 16;  // 4000

    k_enc_ln<<<rowBlocks, 512, 0, stream>>>(y1, enc_w, gamma, beta, enc, normed);
    k_g1<<<dim3(rowBlocks, 8), 128, 0, stream>>>(normed, Wih1, bih1, bhh1, G1);
    k_pack_wih2<<<32, 512, 0, stream>>>(Wih2, wih2f);
    k_lstm<<<1, 512, 311296, stream>>>(G1, wih2f, Whh1, Whh2, bih2, bhh2,
                                       inst2, ys2, out_state);
    k_mask<<<dim3(rowBlocks, 4), 128, 0, stream>>>(ys2, Wd, bd, enc, est);
    k_dec<<<dim3(rowBlocks, 8), 128, 0, stream>>>(dec_w, est, decoded);
}
